// Model_30932354466464
// MI455X (gfx1250) — matrix operand order repaired and checked by reading, not compiled, not run on hardware
//
#include <hip/hip_runtime.h>
#include <hip/hip_bf16.h>
#include <stddef.h>

// ---------------- constants from the reference ----------------
#define S_  512
#define T_  128
#define V_  32000
#define E_  512
#define HE_ 256
#define HD_ 512

typedef __attribute__((ext_vector_type(16))) __bf16 v16bf;
typedef __attribute__((ext_vector_type(8)))  float  v8f;

union bf16x16 { v16bf v; unsigned u[8]; };

// Pack two fp32 -> two bf16 (truncation; folds to v_mov_b16 + v_and_or_b32)
static __device__ __forceinline__ unsigned pack_bf2(float a, float b) {
  return (__builtin_bit_cast(unsigned, a) >> 16) |
         (__builtin_bit_cast(unsigned, b) & 0xFFFF0000u);
}

// ---------------------------------------------------------------
// Embedding gather: enc rows 0..511, dec rows 512..639 (shifted tokens)
// ---------------------------------------------------------------
__global__ void gather_embed_k(const int* __restrict__ enc_idx,
                               const int* __restrict__ dec_idx,
                               const int* __restrict__ start,
                               const float* __restrict__ emb,
                               float* __restrict__ enc_emb,
                               float* __restrict__ dec_emb) {
  int i = blockIdx.x * blockDim.x + threadIdx.x;   // (S_+T_)*E_ threads
  int row = i >> 9, col = i & (E_ - 1);
  if (row < S_) {
    enc_emb[(size_t)row * E_ + col] = emb[(size_t)enc_idx[row] * E_ + col];
  } else {
    int t = row - S_;
    int tok = (t == 0) ? start[0] : dec_idx[t - 1];
    dec_emb[(size_t)t * E_ + col] = emb[(size_t)tok * E_ + col];
  }
}

// ---------------------------------------------------------------
// GEMM: C[M,N] = A[M,K] @ B[N,K]^T + bias, fp32 in/out, bf16 WMMA.
// Block tile: 128(M) x 64(N); 4 waves stacked along M share one LDS-staged
// B slab (64 N x 32 K, pre-packed bf16 pairs -> read once from HBM).
// Wave tile: 32x64 -> 8 accumulators, 8 static v_wmma per k-step.
// Requires M%128==0, N%64==0, K%32==0.
// ---------------------------------------------------------------
__global__ void gemm_bf16_wmma_k(const float* __restrict__ A, int lda,
                                 const float* __restrict__ B, int ldb,
                                 const float* __restrict__ bias,
                                 float* __restrict__ C, int ldc,
                                 int M, int N, int K) {
  __shared__ unsigned Bbuf[4][8][32];            // [nt][v][lane] packed bf16x2, 4 KB
  const int tid  = threadIdx.x;                  // 128
  const int wave = tid >> 5;
  const int lane = tid & 31;
  const int n0 = blockIdx.x * 64;                // block covers N [n0, n0+64)
  const int m0 = blockIdx.y * 128 + wave * 32;   // wave covers M [m0, m0+32)
  const int m     = lane & 15;
  const int kbase = (lane >> 4) << 3;            // 0 | 8 per half-wave

  v8f acc[2][4] = {};
  for (int kb = 0; kb < K; kb += 32) {
    // ---- cooperative B slab fill: 64 rows x 32 k, packed to bf16 pairs ----
#pragma unroll
    for (int e = 0; e < 8; ++e) {
      const int i  = e * 128 + tid;              // 0..1023
      const int v  = (i >> 5) & 7;
      const int k  = ((i >> 4) & 1) * 8 + 2 * (v & 3) + ((v & 4) ? 16 : 0);
      const int nt = i >> 8;
      const float* p = B + (size_t)(n0 + nt * 16 + (i & 15)) * ldb + kb + k;
      if (kb + 32 < K) __builtin_prefetch(p + 32, 0, 1);   // global_prefetch_b8
      Bbuf[nt][v][i & 31] = pack_bf2(p[0], p[1]);
    }
    // ---- two A fragments per wave (rows m0..m0+15, m0+16..m0+31) ----
    bf16x16 a0, a1;
    const float* Ar0 = A + (size_t)(m0 + m) * lda + kb + kbase;
    const float* Ar1 = Ar0 + (size_t)16 * lda;
#pragma unroll
    for (int v = 0; v < 8; ++v) {
      const int k0 = 2 * (v & 3) + ((v & 4) ? 16 : 0);
      a0.u[v] = pack_bf2(Ar0[k0], Ar0[k0 + 1]);
      a1.u[v] = pack_bf2(Ar1[k0], Ar1[k0 + 1]);
    }
    __syncthreads();
    // ---- consume: B fragment from LDS (last index = lane = 16*half + n; conflict-free per v) ----
#pragma unroll
    for (int nt = 0; nt < 4; ++nt) {
      bf16x16 bf;
#pragma unroll
      for (int v = 0; v < 8; ++v) bf.u[v] = Bbuf[nt][v][lane];
      acc[0][nt] = __builtin_amdgcn_wmma_f32_16x16x32_bf16(
          false, a0.v, false, bf.v, (short)0, acc[0][nt], false, false);
      acc[1][nt] = __builtin_amdgcn_wmma_f32_16x16x32_bf16(
          false, a1.v, false, bf.v, (short)0, acc[1][nt], false, false);
    }
    __syncthreads();
  }
  // ---- writeback: VGPR r -> M = mb + r, N = n0 + nt*16 + (lane&15) ----
  const int mlo = (lane >> 4) << 3;
#pragma unroll
  for (int mt = 0; mt < 2; ++mt) {
    const int mb = m0 + mt * 16 + mlo;
#pragma unroll
    for (int nt = 0; nt < 4; ++nt) {
      const int cn = n0 + nt * 16 + (lane & 15);
      const float bv = bias[cn];
#pragma unroll
      for (int r = 0; r < 8; ++r) {
        C[(size_t)(mb + r) * ldc + cn] = acc[mt][nt][r] + bv;
      }
    }
  }
}

// ---------------------------------------------------------------
// Encoder bidirectional LSTM scan. grid.x = 2 (dir), 1024 threads.
// ---------------------------------------------------------------
__global__ void enc_scan_k(const float* __restrict__ Gf,
                           const float* __restrict__ Gb,
                           const float* __restrict__ Whh_f,
                           const float* __restrict__ Whh_b,
                           float* __restrict__ lstm_out) {
  const int dir = blockIdx.x;
  const float* G = dir ? Gb : Gf;
  const float* W = dir ? Whh_b : Whh_f;
  __shared__ float h[HE_], c[HE_], gates[4 * HE_];
  const int tid = threadIdx.x;                 // 1024
  if (tid < HE_) { h[tid] = 0.f; c[tid] = 0.f; }
  __syncthreads();
  for (int s = 0; s < S_; ++s) {
    const int t = dir ? (S_ - 1 - s) : s;
    float a = G[(size_t)t * (4 * HE_) + tid];
    const float* wr = W + (size_t)tid * HE_;
    for (int k = 0; k < HE_; ++k) a = fmaf(wr[k], h[k], a);
    gates[tid] = a;
    __syncthreads();
    if (tid < HE_) {
      float ig = gates[tid], fg = gates[HE_ + tid];
      float gg = gates[2 * HE_ + tid], og = gates[3 * HE_ + tid];
      float ci = 1.f / (1.f + __expf(-ig));
      float cf = 1.f / (1.f + __expf(-fg));
      float co = 1.f / (1.f + __expf(-og));
      float cn = cf * c[tid] + ci * tanhf(gg);
      float hn = co * tanhf(cn);
      c[tid] = cn; h[tid] = hn;
      lstm_out[(size_t)t * HD_ + dir * HE_ + tid] = hn;
    }
    __syncthreads();
  }
}

// ---------------------------------------------------------------
// Decoder scan: LSTM + attention (Pbase-precomputed) + context.
// ---------------------------------------------------------------
__global__ void dec_scan_k(const float* __restrict__ Gd,      // [T_, 2048]
                           const float* __restrict__ Whh,     // [2048, 512]
                           const float* __restrict__ lstm_out,// [S_, 512]
                           const float* __restrict__ Pbase,   // [S_, 256]
                           const float* __restrict__ aW1,     // [256, 1024]
                           const float* __restrict__ aW2,     // [256]
                           const float* __restrict__ ab2,     // [1]
                           float* __restrict__ X,             // [T_, 1024]
                           float* __restrict__ cov,           // [T_, S_]
                           float* __restrict__ attn) {        // [T_, S_]
  __shared__ float h[HD_], c[HD_], gates[4 * HD_];
  __shared__ float qv[256], sc[S_], w[S_], red[32];
  __shared__ float smax, ssum, acum;
  const int tid = threadIdx.x;                 // 1024
  if (tid < HD_) { h[tid] = lstm_out[(size_t)(S_ - 1) * HD_ + tid]; c[tid] = 0.f; }
  if (tid == 0) acum = 0.f;
  __syncthreads();
  for (int t = 0; t < T_; ++t) {
#pragma unroll
    for (int rr = 0; rr < 2; ++rr) {
      const int row = tid + rr * 1024;
      float a = Gd[(size_t)t * (4 * HD_) + row];
      const float* wr = Whh + (size_t)row * HD_;
      for (int k = 0; k < HD_; ++k) a = fmaf(wr[k], h[k], a);
      gates[row] = a;
    }
    __syncthreads();
    if (tid < HD_) {
      float ig = gates[tid], fg = gates[HD_ + tid];
      float gg = gates[2 * HD_ + tid], og = gates[3 * HD_ + tid];
      float ci = 1.f / (1.f + __expf(-ig));
      float cf = 1.f / (1.f + __expf(-fg));
      float co = 1.f / (1.f + __expf(-og));
      float cn = cf * c[tid] + ci * tanhf(gg);
      float hn = co * tanhf(cn);
      c[tid] = cn; h[tid] = hn;
    }
    __syncthreads();
    if (tid < 256) {
      float a = 0.f;
      const float* wr = aW1 + (size_t)tid * (2 * HD_) + HD_;
      for (int k = 0; k < HD_; ++k) a = fmaf(wr[k], h[k], a);
      qv[tid] = a;
    }
    __syncthreads();
    if (tid < S_) {
      float a = ab2[0];
      const float* pb = Pbase + (size_t)tid * 256;
      for (int j = 0; j < 256; ++j) {
        float v = fmaxf(pb[j] + qv[j], 0.f);
        a = fmaf(aW2[j], v, a);
      }
      sc[tid] = fmaxf(a, 0.f);
    }
    __syncthreads();
    if (tid < 32) {
      float mx = -1e30f;
      for (int j = 0; j < 16; ++j) mx = fmaxf(mx, sc[tid * 16 + j]);
      red[tid] = mx;
    }
    __syncthreads();
    if (tid == 0) {
      float mx = red[0];
      for (int j = 1; j < 32; ++j) mx = fmaxf(mx, red[j]);
      smax = mx;
    }
    __syncthreads();
    if (tid < S_) w[tid] = __expf(sc[tid] - smax);
    __syncthreads();
    if (tid < 32) {
      float a = 0.f;
      for (int j = 0; j < 16; ++j) a += w[tid * 16 + j];
      red[tid] = a;
    }
    __syncthreads();
    if (tid == 0) {
      float a = 0.f;
      for (int j = 0; j < 32; ++j) a += red[j];
      ssum = a;
    }
    __syncthreads();
    const float inv = 1.f / ssum;
    if (tid < S_) w[tid] *= inv;
    __syncthreads();
    if (tid < S_) {
      const float* lr = lstm_out + (size_t)tid * HD_;
      float ctx = 0.f;
      for (int j = 0; j < S_; ++j) ctx = fmaf(lr[j], w[j], ctx);
      X[(size_t)t * (2 * HD_) + tid]       = ctx;
      X[(size_t)t * (2 * HD_) + HD_ + tid] = h[tid];
      attn[(size_t)t * S_ + tid] = w[tid];
      cov[(size_t)t * S_ + tid]  = acum;
    }
    __syncthreads();
    if (tid == 0) acum += 1.0f;
    __syncthreads();
  }
}

// ---------------------------------------------------------------
extern "C" void kernel_launch(void* const* d_in, const int* in_sizes, int n_in,
                              void* d_out, int out_size, void* d_ws, size_t ws_size,
                              hipStream_t stream) {
  (void)in_sizes; (void)n_in; (void)out_size; (void)ws_size;
  const int*   enc_idx = (const int*)  d_in[0];
  const int*   dec_idx = (const int*)  d_in[1];
  const int*   start   = (const int*)  d_in[2];
  /* d_in[3] = isTrain (unused) */
  const float* emb     = (const float*)d_in[4];
  const float* ef_Wih  = (const float*)d_in[5];
  const float* ef_Whh  = (const float*)d_in[6];
  const float* ef_b    = (const float*)d_in[7];
  const float* eb_Wih  = (const float*)d_in[8];
  const float* eb_Whh  = (const float*)d_in[9];
  const float* eb_b    = (const float*)d_in[10];
  const float* d_Wih   = (const float*)d_in[11];
  const float* d_Whh   = (const float*)d_in[12];
  const float* d_b     = (const float*)d_in[13];
  const float* aW1     = (const float*)d_in[14];
  const float* ab1     = (const float*)d_in[15];
  const float* aW2     = (const float*)d_in[16];
  const float* ab2     = (const float*)d_in[17];
  const float* oW      = (const float*)d_in[18];
  const float* ob      = (const float*)d_in[19];

  // workspace layout (floats)
  float* ws       = (float*)d_ws;
  float* enc_emb  = ws;                                  // [512,512]
  float* dec_emb  = enc_emb  + (size_t)S_ * E_;          // [128,512]
  float* Gf       = dec_emb  + (size_t)T_ * E_;          // [512,1024]
  float* Gb       = Gf       + (size_t)S_ * 4 * HE_;     // [512,1024]
  float* Gd       = Gb       + (size_t)S_ * 4 * HE_;     // [128,2048]
  float* lstm_out = Gd       + (size_t)T_ * 4 * HD_;     // [512,512]
  float* Pbase    = lstm_out + (size_t)S_ * HD_;         // [512,256]
  float* X        = Pbase    + (size_t)S_ * 256;         // [128,1024]

  // outputs: logits [T,V], coverage [T,S], attn [T,S]
  float* logits = (float*)d_out;
  float* cov    = logits + (size_t)T_ * V_;
  float* attn   = cov    + (size_t)T_ * S_;

  // 1) gather embeddings
  {
    int total = (S_ + T_) * E_;
    gather_embed_k<<<total / 256, 256, 0, stream>>>(enc_idx, dec_idx, start,
                                                    emb, enc_emb, dec_emb);
  }
  // 2) batched input-gate GEMMs (bf16 WMMA)
  gemm_bf16_wmma_k<<<dim3((4 * HE_) / 64, S_ / 128), 128, 0, stream>>>(
      enc_emb, E_, ef_Wih, E_, ef_b, Gf, 4 * HE_, S_, 4 * HE_, E_);
  gemm_bf16_wmma_k<<<dim3((4 * HE_) / 64, S_ / 128), 128, 0, stream>>>(
      enc_emb, E_, eb_Wih, E_, eb_b, Gb, 4 * HE_, S_, 4 * HE_, E_);
  gemm_bf16_wmma_k<<<dim3((4 * HD_) / 64, T_ / 128), 128, 0, stream>>>(
      dec_emb, E_, d_Wih, E_, d_b, Gd, 4 * HD_, T_, 4 * HD_, E_);
  // 3) encoder scan (both directions concurrently)
  enc_scan_k<<<2, 1024, 0, stream>>>(Gf, Gb, ef_Whh, eb_Whh, lstm_out);
  // 4) attention base: Pbase = lstm_out @ aW1[:, :512]^T + ab1  (no relu yet)
  gemm_bf16_wmma_k<<<dim3(256 / 64, S_ / 128), 128, 0, stream>>>(
      lstm_out, HD_, aW1, 2 * HD_, ab1, Pbase, 256, S_, 256, HD_);
  // 5) decoder scan
  dec_scan_k<<<1, 1024, 0, stream>>>(Gd, d_Whh, lstm_out, Pbase,
                                     aW1, aW2, ab2, X, cov, attn);
  // 6) logits = X @ oW^T + ob : oW streamed from HBM exactly once (grid.y == 1)
  gemm_bf16_wmma_k<<<dim3(V_ / 64, T_ / 128), 128, 0, stream>>>(
      X, 2 * HD_, oW, 2 * HD_, ob, logits, V_, T_, V_, 2 * HD_);
}
